// Mamba2Block_53068615909682
// MI455X (gfx1250) — matrix operand order repaired and checked by reading, not compiled, not run on hardware
//
#include <hip/hip_runtime.h>

// ---------------- problem constants (match reference) ----------------
#define D_MODEL 1024
#define D_INNER 2048
#define D_STATE 16
#define D_CONV 4
#define DT_RANK 64
#define BATCH 2
#define SEQ 4096
#define NROWS (BATCH * SEQ)                        // 8192
#define XZ_W (2 * D_INNER + DT_RANK + 2 * D_STATE) // 4192
#define XZ_WPAD 4224                               // next multiple of 128
#define COL_Z 0
#define COL_X D_INNER                              // 2048
#define COL_DT (2 * D_INNER)                       // 4096
#define COL_Bm (2 * D_INNER + DT_RANK)             // 4160
#define COL_Cm (COL_Bm + D_STATE)                  // 4176
#define CHUNK 128
#define NCH (SEQ / CHUNK)                          // 32
#define EPSF 1e-6f

typedef __attribute__((ext_vector_type(16))) __bf16 bf16x16;
typedef __attribute__((ext_vector_type(8)))  float  f32x8;
typedef __attribute__((ext_vector_type(4)))  int    i32x4;
typedef unsigned short u16_t;

static_assert(sizeof(bf16x16) == 32, "bf16x16 must be 32 bytes");

union FragAB { unsigned u[8]; bf16x16 v; };
union FragC  { f32x8 v; float f[8]; };

#define AS1 __attribute__((address_space(1)))
#define AS3 __attribute__((address_space(3)))

#if defined(__gfx1250__) && __has_builtin(__builtin_amdgcn_global_load_async_to_lds_b128)
#define HAVE_ASYNC_LDS 1
#else
#define HAVE_ASYNC_LDS 0
#endif

__device__ __forceinline__ void async_cp16(const void* g, void* l) {
#if HAVE_ASYNC_LDS
  __builtin_amdgcn_global_load_async_to_lds_b128(
      (AS1 i32x4*)g, (AS3 i32x4*)l, 0, 0);
#endif
}

__device__ __forceinline__ void wait_async0() {
#if HAVE_ASYNC_LDS
#if __has_builtin(__builtin_amdgcn_s_wait_asynccnt)
  __builtin_amdgcn_s_wait_asynccnt(0);
#else
  asm volatile("s_wait_asynccnt 0" ::: "memory");
#endif
#endif
}

__device__ __forceinline__ u16_t f32_to_bf16(float f) {
  unsigned u = __float_as_uint(f);
  u += 0x7FFFu + ((u >> 16) & 1u);  // round-to-nearest-even
  return (u16_t)(u >> 16);
}

__device__ __forceinline__ float silu_f(float x) { return x / (1.f + expf(-x)); }

// ---------------- converters ----------------
__global__ __launch_bounds__(256) void cvt_f32_bf16_kernel(
    const float* __restrict__ in, u16_t* __restrict__ out, int n) {
  int i = blockIdx.x * 256 + threadIdx.x;
  if (i < n) out[i] = f32_to_bf16(in[i]);
}

// weight transpose+convert: in [K x N] f32 -> out [Npad x K] bf16 (rows >= N zeroed)
__global__ __launch_bounds__(256) void cvt_wT_bf16_kernel(
    const float* __restrict__ in, u16_t* __restrict__ out, int K, int N, int Npad) {
  int i = blockIdx.x * 256 + threadIdx.x;
  if (i >= Npad * K) return;
  int n = i / K, k = i - n * K;
  float v = (n < N) ? in[(size_t)k * N + n] : 0.f;
  out[i] = f32_to_bf16(v);
}

// dt_raw columns of xz -> compact bf16 (8192 x 64)
__global__ __launch_bounds__(256) void cvt_dtraw_kernel(
    const float* __restrict__ xz, u16_t* __restrict__ out) {
  int i = blockIdx.x * 256 + threadIdx.x;
  if (i >= NROWS * DT_RANK) return;
  int r = i >> 6, c = i & 63;
  out[i] = f32_to_bf16(xz[(size_t)r * XZ_W + COL_DT + c]);
}

// ---------------- RMSNorm -> bf16 ----------------
__global__ __launch_bounds__(256) void rmsnorm_kernel(
    const float* __restrict__ h, const float* __restrict__ w, u16_t* __restrict__ out) {
  __shared__ float red[256];
  int row = blockIdx.x;
  const float* hr = h + (size_t)row * D_MODEL;
  float s = 0.f;
  for (int i = threadIdx.x; i < D_MODEL; i += 256) { float v = hr[i]; s += v * v; }
  red[threadIdx.x] = s;
  __syncthreads();
  for (int off = 128; off > 0; off >>= 1) {
    if (threadIdx.x < off) red[threadIdx.x] += red[threadIdx.x + off];
    __syncthreads();
  }
  float inv = rsqrtf(red[0] / (float)D_MODEL + EPSF);
  for (int i = threadIdx.x; i < D_MODEL; i += 256)
    out[(size_t)row * D_MODEL + i] = f32_to_bf16(hr[i] * inv * w[i]);
}

// ---------------- bf16 WMMA GEMM  C(MxN) = A(MxK) * Bt(NxK)^T ----------------
// A: [M x K] bf16 row-major; Bt: [Npad x K] bf16 (pre-transposed weights)
// mode 0: store; mode 1: softplus(acc + aux[col]); mode 2: acc + aux[row*N+col]
#define BLK_M 128
#define BLK_N 128
#define BLK_K 64
#define LDT 72   // BLK_K + 8 pad (bf16 elems)

__global__ __launch_bounds__(256) void gemm_bf16_wmma_kernel(
    const u16_t* __restrict__ A, const u16_t* __restrict__ Bt, float* __restrict__ C,
    int M, int N, int K, int mode, const float* __restrict__ aux) {
  __shared__ u16_t As[2][BLK_M][LDT];   // [buf][m][k]
  __shared__ u16_t Bs[2][BLK_N][LDT];   // [buf][n][k]
  const int tid   = threadIdx.x;
  const int lane  = tid & 31;
  const int wave  = tid >> 5;
  const int wm    = wave & 3;        // 4 waves along M
  const int wn    = wave >> 2;       // 2 waves along N
  const int half  = lane >> 4;
  const int lan16 = lane & 15;
  const int m0 = blockIdx.y * BLK_M;
  const int n0 = blockIdx.x * BLK_N;

  FragC acc[2][4];
#pragma unroll
  for (int mf = 0; mf < 2; ++mf)
#pragma unroll
    for (int nf = 0; nf < 4; ++nf)
#pragma unroll
      for (int r = 0; r < 8; ++r) acc[mf][nf].f[r] = 0.f;

  // tile loads: 64 contiguous bytes per thread for each of A and B
  const int trow = tid >> 1, thalf = tid & 1;
  const u16_t* gA = A  + (size_t)(m0 + trow) * K + thalf * 32;
  const u16_t* gB = Bt + (size_t)(n0 + trow) * K + thalf * 32;

  auto load_tile = [&](int buf, int kb) {
#if HAVE_ASYNC_LDS
#pragma unroll
    for (int q = 0; q < 4; ++q) {
      async_cp16(gA + kb + q * 8, &As[buf][trow][thalf * 32 + q * 8]);
      async_cp16(gB + kb + q * 8, &Bs[buf][trow][thalf * 32 + q * 8]);
    }
#else
    const uint4* ga = reinterpret_cast<const uint4*>(gA + kb);
    uint4* la = reinterpret_cast<uint4*>(&As[buf][trow][thalf * 32]);
    la[0] = ga[0]; la[1] = ga[1]; la[2] = ga[2]; la[3] = ga[3];
    const uint4* gb = reinterpret_cast<const uint4*>(gB + kb);
    uint4* lb = reinterpret_cast<uint4*>(&Bs[buf][trow][thalf * 32]);
    lb[0] = gb[0]; lb[1] = gb[1]; lb[2] = gb[2]; lb[3] = gb[3];
    if (kb + BLK_K < K) {
      __builtin_prefetch(gA + kb + BLK_K, 0, 0);
      __builtin_prefetch(gB + kb + BLK_K, 0, 0);
    }
#endif
  };

  // prologue: stage tile 0
  load_tile(0, 0);
  wait_async0();
  __syncthreads();

  int buf = 0;
  for (int kb = 0; kb < K; kb += BLK_K) {
    // stream next tile into the other buffer while computing this one
    if (kb + BLK_K < K) load_tile(buf ^ 1, kb + BLK_K);

#pragma unroll
    for (int kk = 0; kk < BLK_K; kk += 32) {
      FragAB af[2], bfr[4];
      // A fragment: ISA 16-bit A layout (16x32): k = (i>>2)*16 + half*8 + (i&3)*2
#pragma unroll
      for (int mf = 0; mf < 2; ++mf) {
        const u16_t* base = &As[buf][wm * 32 + mf * 16 + lan16][kk];
#pragma unroll
        for (int i = 0; i < 8; ++i) {
          int k = ((i >> 2) << 4) + (half << 3) + ((i & 3) << 1);
          af[mf].u[i] = *reinterpret_cast<const unsigned*>(base + k);
        }
      }
      // B fragment: same k order as A: k = (i>>2)*16 + half*8 + (i&3)*2
#pragma unroll
      for (int nf = 0; nf < 4; ++nf) {
        const u16_t* base = &Bs[buf][wn * 64 + nf * 16 + lan16][kk];
#pragma unroll
        for (int i = 0; i < 8; ++i) {
          int k = ((i >> 2) << 4) + (half << 3) + ((i & 3) << 1);
          bfr[nf].u[i] = *reinterpret_cast<const unsigned*>(base + k);
        }
      }
#pragma unroll
      for (int mf = 0; mf < 2; ++mf)
#pragma unroll
        for (int nf = 0; nf < 4; ++nf)
          acc[mf][nf].v = __builtin_amdgcn_wmma_f32_16x16x32_bf16(
              false, af[mf].v, false, bfr[nf].v, (short)0, acc[mf][nf].v, false, false);
    }

    wait_async0();     // my async stores into buf^1 landed
    __syncthreads();   // everyone done reading buf & done staging buf^1
    buf ^= 1;
  }

  // epilogue: C/D layout -> row = base + half*8 + r, col = lan16
#pragma unroll
  for (int mf = 0; mf < 2; ++mf) {
#pragma unroll
    for (int nf = 0; nf < 4; ++nf) {
      int col = n0 + wn * 64 + nf * 16 + lan16;
      if (col >= N) continue;
#pragma unroll
      for (int r = 0; r < 8; ++r) {
        int row = m0 + wm * 32 + mf * 16 + half * 8 + r;
        size_t o = (size_t)row * N + col;
        float v = acc[mf][nf].f[r];
        if (mode == 1) { v += aux[col]; v = (v > 20.f) ? v : log1pf(expf(v)); }
        else if (mode == 2) { v += aux[o]; }
        C[o] = v;
      }
    }
  }
}

// ---------------- causal depthwise conv (K=4) + bias + SiLU ----------------
__global__ __launch_bounds__(256) void conv_silu_kernel(
    const float* __restrict__ xz, const float* __restrict__ cw,
    const float* __restrict__ cb, float* __restrict__ xconv) {
  int i = blockIdx.x * 256 + threadIdx.x;
  if (i >= NROWS * D_INNER) return;
  int row = i >> 11;            // /D_INNER
  int d   = i & (D_INNER - 1);
  int l   = row & (SEQ - 1);
  float acc = cb[d];
#pragma unroll
  for (int j = 0; j < D_CONV; ++j) {
    int lt = l - (D_CONV - 1) + j;
    if (lt >= 0)
      acc += cw[d * D_CONV + j] * xz[(size_t)(row - (D_CONV - 1) + j) * XZ_W + COL_X + d];
  }
  xconv[i] = silu_f(acc);
}

// ---------------- chunked selective scan ----------------
// Phase 1: per-chunk local scan from h=0; emit end-state S and sum(dt)
__global__ __launch_bounds__(256) void scan_local_kernel(
    const float* __restrict__ xz, const float* __restrict__ dtf,
    const float* __restrict__ xc, const float* __restrict__ A_log,
    float* __restrict__ S, float* __restrict__ sumdt) {
  __shared__ float Bsh[CHUNK * D_STATE];
  int b = blockIdx.z, ch = blockIdx.y;
  int d = blockIdx.x * 256 + threadIdx.x;
  for (int idx = threadIdx.x; idx < CHUNK * D_STATE; idx += 256) {
    int t = idx >> 4, n = idx & 15;
    Bsh[idx] = xz[(size_t)(b * SEQ + ch * CHUNK + t) * XZ_W + COL_Bm + n];
  }
  __syncthreads();
  float a[D_STATE], h[D_STATE];
#pragma unroll
  for (int n = 0; n < D_STATE; ++n) { a[n] = -expf(A_log[d * D_STATE + n]); h[n] = 0.f; }
  float sd = 0.f;
  for (int t = 0; t < CHUNK; ++t) {
    size_t row = (size_t)b * SEQ + ch * CHUNK + t;
    float dtv = dtf[row * D_INNER + d];
    float xv  = xc[row * D_INNER + d];
    sd += dtv;
    float dx = dtv * xv;
#pragma unroll
    for (int n = 0; n < D_STATE; ++n)
      h[n] = expf(dtv * a[n]) * h[n] + dx * Bsh[t * D_STATE + n];
  }
  size_t ci = ((size_t)(ch * BATCH + b) * D_INNER + d);
#pragma unroll
  for (int n = 0; n < D_STATE; ++n) S[ci * D_STATE + n] = h[n];
  sumdt[ci] = sd;
}

// Phase 2: serial carry over chunks (chunk prefix factor = exp(A * sum(dt)))
__global__ __launch_bounds__(256) void scan_carry_kernel(
    const float* __restrict__ A_log, const float* __restrict__ S,
    const float* __restrict__ sumdt, float* __restrict__ H0) {
  int idx = blockIdx.x * 256 + threadIdx.x;  // 0..BATCH*D_INNER-1
  int b = idx / D_INNER, d = idx & (D_INNER - 1);
  float a[D_STATE], h[D_STATE];
#pragma unroll
  for (int n = 0; n < D_STATE; ++n) { a[n] = -expf(A_log[d * D_STATE + n]); h[n] = 0.f; }
  for (int c = 0; c < NCH; ++c) {
    size_t ci = ((size_t)(c * BATCH + b) * D_INNER + d);
#pragma unroll
    for (int n = 0; n < D_STATE; ++n) H0[ci * D_STATE + n] = h[n];
    float sd = sumdt[ci];
#pragma unroll
    for (int n = 0; n < D_STATE; ++n)
      h[n] = expf(sd * a[n]) * h[n] + S[ci * D_STATE + n];
  }
}

// Phase 3: re-scan with correct chunk start state, emit y*silu(z) as bf16
__global__ __launch_bounds__(256) void scan_final_kernel(
    const float* __restrict__ xz, const float* __restrict__ dtf,
    const float* __restrict__ xc, const float* __restrict__ A_log,
    const float* __restrict__ Dv, const float* __restrict__ H0,
    u16_t* __restrict__ yb) {
  __shared__ float Bsh[CHUNK * D_STATE];
  __shared__ float Csh[CHUNK * D_STATE];
  int b = blockIdx.z, ch = blockIdx.y;
  int d = blockIdx.x * 256 + threadIdx.x;
  for (int idx = threadIdx.x; idx < CHUNK * D_STATE; idx += 256) {
    int t = idx >> 4, n = idx & 15;
    size_t row = (size_t)(b * SEQ + ch * CHUNK + t);
    Bsh[idx] = xz[row * XZ_W + COL_Bm + n];
    Csh[idx] = xz[row * XZ_W + COL_Cm + n];
  }
  __syncthreads();
  float a[D_STATE], h[D_STATE];
  size_t ci = ((size_t)(ch * BATCH + b) * D_INNER + d);
#pragma unroll
  for (int n = 0; n < D_STATE; ++n) {
    a[n] = -expf(A_log[d * D_STATE + n]);
    h[n] = H0[ci * D_STATE + n];
  }
  float Dd = Dv[d];
  for (int t = 0; t < CHUNK; ++t) {
    size_t row = (size_t)b * SEQ + ch * CHUNK + t;
    float dtv = dtf[row * D_INNER + d];
    float xv  = xc[row * D_INNER + d];
    float dx  = dtv * xv;
    float y   = Dd * xv;
#pragma unroll
    for (int n = 0; n < D_STATE; ++n) {
      h[n] = expf(dtv * a[n]) * h[n] + dx * Bsh[t * D_STATE + n];
      y += h[n] * Csh[t * D_STATE + n];
    }
    float z = xz[row * XZ_W + COL_Z + d];
    y *= silu_f(z);
    yb[row * D_INNER + d] = f32_to_bf16(y);
  }
}

// ---------------- host-side orchestration ----------------
extern "C" void kernel_launch(void* const* d_in, const int* in_sizes, int n_in,
                              void* d_out, int out_size, void* d_ws, size_t ws_size,
                              hipStream_t stream) {
  (void)in_sizes; (void)n_in; (void)out_size; (void)ws_size;
  const float* h_in   = (const float*)d_in[0];
  const float* norm_w = (const float*)d_in[1];
  const float* w_in   = (const float*)d_in[2];
  const float* conv_w = (const float*)d_in[3];
  const float* conv_b = (const float*)d_in[4];
  const float* w_dt   = (const float*)d_in[5];
  const float* dt_b   = (const float*)d_in[6];
  const float* A_log  = (const float*)d_in[7];
  const float* Dvec   = (const float*)d_in[8];
  const float* w_out  = (const float*)d_in[9];
  float* out = (float*)d_out;

  char* p = (char*)d_ws;
  auto alloc = [&](size_t bytes) -> void* {
    void* r = (void*)p;
    p += (bytes + 255) & ~(size_t)255;
    return r;
  };
  u16_t* w_in_bT  = (u16_t*)alloc((size_t)XZ_WPAD * D_MODEL * 2);  // [4224][1024]
  u16_t* w_dt_bT  = (u16_t*)alloc((size_t)D_INNER * DT_RANK * 2);  // [2048][64]
  u16_t* w_out_bT = (u16_t*)alloc((size_t)D_MODEL * D_INNER * 2);  // [1024][2048]
  u16_t* hn_b     = (u16_t*)alloc((size_t)NROWS * D_MODEL * 2);
  float* xz       = (float*)alloc((size_t)NROWS * XZ_W * 4);
  u16_t* dtraw_b  = (u16_t*)alloc((size_t)NROWS * DT_RANK * 2);
  float* dt_f     = (float*)alloc((size_t)NROWS * D_INNER * 4);
  float* xconv    = (float*)alloc((size_t)NROWS * D_INNER * 4);
  u16_t* y_b      = (u16_t*)alloc((size_t)NROWS * D_INNER * 2);
  float* carryS   = (float*)alloc((size_t)NCH * BATCH * D_INNER * D_STATE * 4);
  float* sumdt    = (float*)alloc((size_t)NCH * BATCH * D_INNER * 4);
  float* H0       = (float*)alloc((size_t)NCH * BATCH * D_INNER * D_STATE * 4);

  // weight transpose + conversions (f32 [KxN] -> bf16 [NpadxK])
  {
    int n = XZ_WPAD * D_MODEL;
    cvt_wT_bf16_kernel<<<(n + 255) / 256, 256, 0, stream>>>(w_in, w_in_bT, D_MODEL, XZ_W, XZ_WPAD);
    n = D_INNER * DT_RANK;
    cvt_wT_bf16_kernel<<<(n + 255) / 256, 256, 0, stream>>>(w_dt, w_dt_bT, DT_RANK, D_INNER, D_INNER);
    n = D_MODEL * D_INNER;
    cvt_wT_bf16_kernel<<<(n + 255) / 256, 256, 0, stream>>>(w_out, w_out_bT, D_INNER, D_MODEL, D_MODEL);
  }
  // RMSNorm -> bf16 activations
  rmsnorm_kernel<<<NROWS, 256, 0, stream>>>(h_in, norm_w, hn_b);
  // GEMM1: xz = hn @ W_in   (8192 x 4192, K=1024)
  gemm_bf16_wmma_kernel<<<dim3(XZ_WPAD / BLK_N, NROWS / BLK_M), 256, 0, stream>>>(
      hn_b, w_in_bT, xz, NROWS, XZ_W, D_MODEL, 0, nullptr);
  // dt_raw slice -> bf16
  cvt_dtraw_kernel<<<(NROWS * DT_RANK + 255) / 256, 256, 0, stream>>>(xz, dtraw_b);
  // GEMM2: dt = softplus(dt_raw @ W_dt + b)   (8192 x 2048, K=64)
  gemm_bf16_wmma_kernel<<<dim3(D_INNER / BLK_N, NROWS / BLK_M), 256, 0, stream>>>(
      dtraw_b, w_dt_bT, dt_f, NROWS, D_INNER, DT_RANK, 1, dt_b);
  // depthwise conv + SiLU
  conv_silu_kernel<<<(NROWS * D_INNER) / 256, 256, 0, stream>>>(xz, conv_w, conv_b, xconv);
  // chunked selective scan
  scan_local_kernel<<<dim3(D_INNER / 256, NCH, BATCH), 256, 0, stream>>>(
      xz, dt_f, xconv, A_log, carryS, sumdt);
  scan_carry_kernel<<<(BATCH * D_INNER) / 256, 256, 0, stream>>>(A_log, carryS, sumdt, H0);
  scan_final_kernel<<<dim3(D_INNER / 256, NCH, BATCH), 256, 0, stream>>>(
      xz, dt_f, xconv, A_log, Dvec, H0, y_b);
  // GEMM3: out = y @ W_out + residual   (8192 x 1024, K=2048)
  gemm_bf16_wmma_kernel<<<dim3(D_MODEL / BLK_N, NROWS / BLK_M), 256, 0, stream>>>(
      y_b, w_out_bT, out, NROWS, D_MODEL, D_INNER, 2, h_in);
}
